// RULSTM_64604898066623
// MI455X (gfx1250) — matrix operand order repaired and checked by reading, not compiled, not run on hardware
//
#include <hip/hip_runtime.h>

// ---------------------------------------------------------------------------
// RULSTM for MI455X (gfx1250): bf16 WMMA GEMMs + fp32 LSTM cells.
//   roll:   14 x ( h@Whh^T gemm  + cell )   with x@Wih^T hoisted into 1 gemm
//   unroll: 15 x ( h@Whh^T gemm  + cell )   with branch pruning + hoisted x@Wih^T
//   cls:    one [3584,1024]x[1024,2513] gemm straight into d_out
// GEMM inner loop is pure global_load_b128 + v_wmma: the ragged classifier
// N is handled by zero-filling padded cls_W rows once at setup (no in-loop
// masking -> no cndmask/v_nop pollution of the matrix pipe).
// ---------------------------------------------------------------------------

typedef __attribute__((ext_vector_type(16))) __bf16 v16bf;
typedef __attribute__((ext_vector_type(8)))  float  v8f;

#define B_SZ 256
#define T_SZ 14
#define F_SZ 1024
#define H_SZ 1024
#define C_SZ 2513
#define G_SZ (4 * H_SZ)          // 4096 gate width
#define C_PAD 2560               // C rounded up to 256 for safe, zero-filled edge tiles

__device__ __forceinline__ unsigned short f2bf(float x) {
  unsigned int u = __float_as_uint(x);
  u += 0x7FFFu + ((u >> 16) & 1u);            // round-to-nearest-even
  return (unsigned short)(u >> 16);
}
__device__ __forceinline__ float sigmoidf_(float x) { return 1.0f / (1.0f + __expf(-x)); }

struct alignas(32) Frag32B { uint4 a, b; };

__device__ __forceinline__ v16bf load_frag(const unsigned short* p0, const unsigned short* p1) {
  Frag32B f;
  f.a = *(const uint4*)p0;    // elements 0..7
  f.b = *(const uint4*)p1;    // elements 8..15
  return *(v16bf*)&f;
}

// ---------------------------------------------------------------------------
// GEMM: out[M,N] = A[M,K](bf16) @ W[N,K](bf16)^T (+ Cadd[M,N]) (+ bias[N])
// WG = 256 threads = 8 waves (2 over M x 4 over N); WG tile 64(M) x 256(N);
// wave tile 32x64 = 2x4 WMMA accumulators.
// Call-site guarantees: M % 64 == 0, K % 32 == 0, and W has >= gridDim.x*256
// rows of valid (possibly zero-padded) data. Stores are guarded by N.
// ---------------------------------------------------------------------------
__global__ __launch_bounds__(256)
void gemm_bf16_wmma(const unsigned short* __restrict__ A, int lda,
                    const unsigned short* __restrict__ W, int ldw,
                    const float* __restrict__ Cadd, int ldadd,
                    const float* __restrict__ bias,
                    float* __restrict__ out, int ldo,
                    int M, int N, int K)
{
  const int lane = threadIdx.x & 31;
  const int wave = threadIdx.x >> 5;        // 0..7
  const int wm   = wave & 1;                // 2 waves over M
  const int wn   = wave >> 1;               // 4 waves over N
  const int m0   = blockIdx.y * 64  + wm * 32;
  const int n0   = blockIdx.x * 256 + wn * 64;
  if (m0 >= M) return;

  const int  lrow  = lane & 15;
  const bool hi    = lane >= 16;
  const int  abase = hi ? 16 : 0;           // A: lanes>=16 hold K {16..31}
  const int  bkoff = hi ? 16 : 0;           // B: lanes>=16 hold K {16..31}

  v8f acc[2][4] = {};

  const unsigned short* a0p = A + (size_t)(m0 + lrow)      * lda + abase;
  const unsigned short* a1p = A + (size_t)(m0 + 16 + lrow) * lda + abase;
  const unsigned short* bp0 = W + (size_t)(n0 +      lrow) * ldw + bkoff;
  const unsigned short* bp1 = W + (size_t)(n0 + 16 + lrow) * ldw + bkoff;
  const unsigned short* bp2 = W + (size_t)(n0 + 32 + lrow) * ldw + bkoff;
  const unsigned short* bp3 = W + (size_t)(n0 + 48 + lrow) * ldw + bkoff;

  for (int k = 0; k < K; k += 32) {
    v16bf av0 = load_frag(a0p + k, a0p + k + 8);
    v16bf av1 = load_frag(a1p + k, a1p + k + 8);
    v16bf bv0 = load_frag(bp0 + k, bp0 + k + 8);
    v16bf bv1 = load_frag(bp1 + k, bp1 + k + 8);
    v16bf bv2 = load_frag(bp2 + k, bp2 + k + 8);
    v16bf bv3 = load_frag(bp3 + k, bp3 + k + 8);

    acc[0][0] = __builtin_amdgcn_wmma_f32_16x16x32_bf16(false, av0, false, bv0, (short)0, acc[0][0], false, false);
    acc[0][1] = __builtin_amdgcn_wmma_f32_16x16x32_bf16(false, av0, false, bv1, (short)0, acc[0][1], false, false);
    acc[0][2] = __builtin_amdgcn_wmma_f32_16x16x32_bf16(false, av0, false, bv2, (short)0, acc[0][2], false, false);
    acc[0][3] = __builtin_amdgcn_wmma_f32_16x16x32_bf16(false, av0, false, bv3, (short)0, acc[0][3], false, false);
    acc[1][0] = __builtin_amdgcn_wmma_f32_16x16x32_bf16(false, av1, false, bv0, (short)0, acc[1][0], false, false);
    acc[1][1] = __builtin_amdgcn_wmma_f32_16x16x32_bf16(false, av1, false, bv1, (short)0, acc[1][1], false, false);
    acc[1][2] = __builtin_amdgcn_wmma_f32_16x16x32_bf16(false, av1, false, bv2, (short)0, acc[1][2], false, false);
    acc[1][3] = __builtin_amdgcn_wmma_f32_16x16x32_bf16(false, av1, false, bv3, (short)0, acc[1][3], false, false);
  }

  const int rbase = hi ? 8 : 0;             // C/D layout: hi lanes hold rows 8..15
#pragma unroll
  for (int mi = 0; mi < 2; ++mi) {
#pragma unroll
    for (int ni = 0; ni < 4; ++ni) {
      const int ncol = n0 + ni * 16 + lrow;
      if (ncol >= N) continue;
      const float bval = bias ? bias[ncol] : 0.0f;
#pragma unroll
      for (int r = 0; r < 8; ++r) {
        const int mrow = m0 + mi * 16 + rbase + r;
        float v = acc[mi][ni][r] + bval;
        if (Cadd) v += Cadd[(size_t)mrow * ldadd + ncol];
        out[(size_t)mrow * ldo + ncol] = v;
      }
    }
  }
}

// --------------------------- elementwise helpers ---------------------------
__global__ void k_f2bf(const float* __restrict__ in, unsigned short* __restrict__ out, int n) {
  int i = blockIdx.x * blockDim.x + threadIdx.x;
  if (i < n) out[i] = f2bf(in[i]);
}
__global__ void k_zero_u16(unsigned short* __restrict__ out, int n) {
  int i = blockIdx.x * blockDim.x + threadIdx.x;
  if (i < n) out[i] = 0;
}
__global__ void k_addv(const float* __restrict__ a, const float* __restrict__ b,
                       float* __restrict__ o, int n) {
  int i = blockIdx.x * blockDim.x + threadIdx.x;
  if (i < n) o[i] = a[i] + b[i];
}
// inputs [B,T,F] fp32 -> x_bf [T,B,F] bf16
__global__ void k_xpose_bf(const float* __restrict__ in, unsigned short* __restrict__ out, int n) {
  int idx = blockIdx.x * blockDim.x + threadIdx.x;
  if (idx >= n) return;
  int f  = idx % F_SZ;
  int rb = idx / F_SZ;                 // t*B + b
  int t  = rb / B_SZ;
  int b  = rb % B_SZ;
  out[idx] = f2bf(in[((size_t)b * T_SZ + t) * F_SZ + f]);
}
__global__ void k_init_state(float* __restrict__ h, float* __restrict__ c,
                             unsigned short* __restrict__ hbf, int n) {
  int i = blockIdx.x * blockDim.x + threadIdx.x;
  if (i < n) { h[i] = 0.0f; c[i] = 0.0f; hbf[i] = 0; }
}

// Roll cell: update (h,c), emit bf16 h for next gemm, and seed unroll state slot t.
__global__ void k_cell_roll(const float* __restrict__ gates, float* __restrict__ h,
                            float* __restrict__ c, unsigned short* __restrict__ hbf,
                            float* __restrict__ hu, float* __restrict__ cu,
                            unsigned short* __restrict__ hubf, int n) {
  int idx = blockIdx.x * blockDim.x + threadIdx.x;
  if (idx >= n) return;
  int b = idx / H_SZ, j = idx % H_SZ;
  const float* g = gates + (size_t)b * G_SZ;
  float ig = sigmoidf_(g[j]);
  float fg = sigmoidf_(g[H_SZ + j]);
  float gg = tanhf(g[2 * H_SZ + j]);
  float og = sigmoidf_(g[3 * H_SZ + j]);
  float cn = fg * c[idx] + ig * gg;
  float hn = og * tanhf(cn);
  h[idx] = hn;  c[idx] = cn;  hbf[idx] = f2bf(hn);
  hu[idx] = hn; cu[idx] = cn; hubf[idx] = f2bf(hn);
}

// Unroll cell over nrows = nb*B branch rows; branch t_done records its final h
// into hn_bf in [b, t, H] order (classifier / output layout).
__global__ void k_cell_unroll(const float* __restrict__ gates, float* __restrict__ h,
                              float* __restrict__ c, unsigned short* __restrict__ hbf,
                              unsigned short* __restrict__ hnbf, int n, int t_done) {
  int idx = blockIdx.x * blockDim.x + threadIdx.x;
  if (idx >= n) return;
  int row = idx / H_SZ, j = idx % H_SZ;
  const float* g = gates + (size_t)row * G_SZ;
  float ig = sigmoidf_(g[j]);
  float fg = sigmoidf_(g[H_SZ + j]);
  float gg = tanhf(g[2 * H_SZ + j]);
  float og = sigmoidf_(g[3 * H_SZ + j]);
  float cn = fg * c[idx] + ig * gg;
  float hn = og * tanhf(cn);
  h[idx] = hn; c[idx] = cn; hbf[idx] = f2bf(hn);
  if (t_done >= 0 && row >= t_done * B_SZ) {
    int b = row - t_done * B_SZ;
    hnbf[((size_t)b * T_SZ + t_done) * H_SZ + j] = f2bf(hn);
  }
}

// ---------------------------------------------------------------------------
extern "C" void kernel_launch(void* const* d_in, const int* in_sizes, int n_in,
                              void* d_out, int out_size, void* d_ws, size_t ws_size,
                              hipStream_t stream) {
  (void)in_sizes; (void)n_in; (void)out_size; (void)ws_size;
  const float* inputs = (const float*)d_in[0];
  const float* rWih   = (const float*)d_in[1];
  const float* rWhh   = (const float*)d_in[2];
  const float* rbih   = (const float*)d_in[3];
  const float* rbhh   = (const float*)d_in[4];
  const float* uWih   = (const float*)d_in[5];
  const float* uWhh   = (const float*)d_in[6];
  const float* ubih   = (const float*)d_in[7];
  const float* ubhh   = (const float*)d_in[8];
  const float* clsW   = (const float*)d_in[9];
  const float* clsb   = (const float*)d_in[10];
  float* y = (float*)d_out;

  // ---- workspace carve-out (256B aligned) ----
  char* p = (char*)d_ws;
  auto alloc = [&](size_t bytes) -> void* {
    void* r = (void*)p;
    p += (bytes + 255) & ~(size_t)255;
    return r;
  };
  unsigned short* wih_r  = (unsigned short*)alloc((size_t)G_SZ * F_SZ * 2);
  unsigned short* whh_r  = (unsigned short*)alloc((size_t)G_SZ * H_SZ * 2);
  unsigned short* wih_u  = (unsigned short*)alloc((size_t)G_SZ * F_SZ * 2);
  unsigned short* whh_u  = (unsigned short*)alloc((size_t)G_SZ * H_SZ * 2);
  unsigned short* clsw   = (unsigned short*)alloc((size_t)C_PAD * H_SZ * 2); // padded rows
  unsigned short* x_bf   = (unsigned short*)alloc((size_t)T_SZ * B_SZ * F_SZ * 2);
  float*          bsum_r = (float*)alloc((size_t)G_SZ * 4);
  float*          bsum_u = (float*)alloc((size_t)G_SZ * 4);
  float*          xw_r   = (float*)alloc((size_t)T_SZ * B_SZ * G_SZ * 4);   // 56 MB
  float*          xw_u   = (float*)alloc((size_t)T_SZ * B_SZ * G_SZ * 4);   // 56 MB
  float*          gates  = (float*)alloc((size_t)T_SZ * B_SZ * G_SZ * 4);   // 56 MB (reused)
  float*          h_r    = (float*)alloc((size_t)B_SZ * H_SZ * 4);
  float*          c_r    = (float*)alloc((size_t)B_SZ * H_SZ * 4);
  unsigned short* h_r_bf = (unsigned short*)alloc((size_t)B_SZ * H_SZ * 2);
  float*          h_u    = (float*)alloc((size_t)T_SZ * B_SZ * H_SZ * 4);
  float*          c_u    = (float*)alloc((size_t)T_SZ * B_SZ * H_SZ * 4);
  unsigned short* h_u_bf = (unsigned short*)alloc((size_t)T_SZ * B_SZ * H_SZ * 2);
  unsigned short* hn_bf  = (unsigned short*)alloc((size_t)B_SZ * T_SZ * H_SZ * 2);

  const int TB = 256;
  auto blk = [](int n) { return (n + 255) / 256; };

  // ---- one-time conversions / precomputation ----
  k_f2bf<<<blk(G_SZ * F_SZ), TB, 0, stream>>>(rWih, wih_r, G_SZ * F_SZ);
  k_f2bf<<<blk(G_SZ * H_SZ), TB, 0, stream>>>(rWhh, whh_r, G_SZ * H_SZ);
  k_f2bf<<<blk(G_SZ * F_SZ), TB, 0, stream>>>(uWih, wih_u, G_SZ * F_SZ);
  k_f2bf<<<blk(G_SZ * H_SZ), TB, 0, stream>>>(uWhh, whh_u, G_SZ * H_SZ);
  k_f2bf<<<blk(C_SZ * H_SZ), TB, 0, stream>>>(clsW, clsw, C_SZ * H_SZ);
  k_zero_u16<<<blk((C_PAD - C_SZ) * H_SZ), TB, 0, stream>>>(
      clsw + (size_t)C_SZ * H_SZ, (C_PAD - C_SZ) * H_SZ);      // zero pad rows
  k_addv<<<blk(G_SZ), TB, 0, stream>>>(rbih, rbhh, bsum_r, G_SZ);
  k_addv<<<blk(G_SZ), TB, 0, stream>>>(ubih, ubhh, bsum_u, G_SZ);
  k_xpose_bf<<<blk(T_SZ * B_SZ * F_SZ), TB, 0, stream>>>(inputs, x_bf, T_SZ * B_SZ * F_SZ);
  k_init_state<<<blk(B_SZ * H_SZ), TB, 0, stream>>>(h_r, c_r, h_r_bf, B_SZ * H_SZ);

  // Hoisted x@Wih^T + (bih+bhh) for both LSTMs: [T*B, 4H]
  {
    dim3 grid(G_SZ / 256, (T_SZ * B_SZ) / 64);
    gemm_bf16_wmma<<<grid, TB, 0, stream>>>(x_bf, F_SZ, wih_r, F_SZ, nullptr, 0,
                                            bsum_r, xw_r, G_SZ, T_SZ * B_SZ, G_SZ, F_SZ);
    gemm_bf16_wmma<<<grid, TB, 0, stream>>>(x_bf, F_SZ, wih_u, F_SZ, nullptr, 0,
                                            bsum_u, xw_u, G_SZ, T_SZ * B_SZ, G_SZ, F_SZ);
  }

  // ---- rolling LSTM: 14 sequential steps ----
  for (int t = 0; t < T_SZ; ++t) {
    dim3 grid(G_SZ / 256, B_SZ / 64);
    gemm_bf16_wmma<<<grid, TB, 0, stream>>>(h_r_bf, H_SZ, whh_r, H_SZ,
                                            xw_r + (size_t)t * B_SZ * G_SZ, G_SZ,
                                            nullptr, gates, G_SZ, B_SZ, G_SZ, H_SZ);
    k_cell_roll<<<blk(B_SZ * H_SZ), TB, 0, stream>>>(
        gates, h_r, c_r, h_r_bf,
        h_u + (size_t)t * B_SZ * H_SZ, c_u + (size_t)t * B_SZ * H_SZ,
        h_u_bf + (size_t)t * B_SZ * H_SZ, B_SZ * H_SZ);
  }

  // ---- unrolling LSTM: T+1 steps, branches pruned as they finish ----
  for (int k = 0; k <= T_SZ; ++k) {
    int nb = T_SZ - k + 1;
    if (nb > T_SZ) nb = T_SZ;             // branch t active iff t <= T-k
    const int M = nb * B_SZ;              // always a multiple of 64
    const int t_done = (k >= 1) ? (T_SZ - k) : -1;
    dim3 grid(G_SZ / 256, M / 64);
    gemm_bf16_wmma<<<grid, TB, 0, stream>>>(h_u_bf, H_SZ, whh_u, H_SZ,
                                            xw_u, G_SZ, nullptr, gates, G_SZ,
                                            M, G_SZ, H_SZ);
    k_cell_unroll<<<blk(M * H_SZ), TB, 0, stream>>>(gates, h_u, c_u, h_u_bf,
                                                    hn_bf, M * H_SZ, t_done);
  }

  // ---- classifier: y[B*T, C] = hn @ cls_W^T + cls_b (fp32 out) ----
  {
    dim3 grid(C_PAD / 256, (B_SZ * T_SZ) / 64);
    gemm_bf16_wmma<<<grid, TB, 0, stream>>>(hn_bf, H_SZ, clsw, H_SZ, nullptr, 0,
                                            clsb, y, C_SZ, B_SZ * T_SZ, C_SZ, H_SZ);
  }
}
